// TENFinalLayer_7017976561853
// MI455X (gfx1250) — matrix operand order repaired and checked by reading, not compiled, not run on hardware
//
#include <hip/hip_runtime.h>

#define NTOK  16384   // B*T
#define DDIM  1024
#define TWOK  128
#define KCH   64
#define MLPD  4096
#define SEQT  4096
#define NB    4

typedef __attribute__((ext_vector_type(16))) __bf16 v16bf;
typedef __attribute__((ext_vector_type(8)))  float  v8f;

union Frag { v16bf v; uint4 u[2]; };

// ---------------------------------------------------------------------------
// A fragment: 16-bit A 16x32 layout (ISA 7.12.2). lane = row (mod 16),
// half=lane>>4 selects K sub-block: elems 0..7 = K[k0+half*8 .. +7],
// elems 8..15 = K[k0+16+half*8 .. +7].
__device__ inline v16bf load_fragA(const __bf16* __restrict__ A, int ld,
                                   int row0, int k0, int lane) {
    const __bf16* p = A + (size_t)(row0 + (lane & 15)) * ld + k0 + ((lane >> 4) << 3);
    Frag f;
    f.u[0] = *reinterpret_cast<const uint4*>(p);
    f.u[1] = *reinterpret_cast<const uint4*>(p + 16);
    return f.v;
}

// B fragment: 16-bit B 32x16 layout. lane = output col (mod 16), half=lane>>4
// selects K sub-block: elems 0..7 = K[k0+half*8 .. +7], elems 8..15 = K[k0+16+half*8 .. +7].
// With W row-major (Nout x K) this computes A * W^T.
__device__ inline v16bf load_fragB(const __bf16* __restrict__ W, int ld,
                                   int col0, int k0, int lane) {
    const __bf16* p = W + (size_t)(col0 + (lane & 15)) * ld + k0 + ((lane >> 4) << 3);
    Frag f;
    f.u[0] = *reinterpret_cast<const uint4*>(p);
    f.u[1] = *reinterpret_cast<const uint4*>(p + 16);
    return f.v;
}

// ---------------------------------------------------------------------------
// Generic GEMM: out = A(bf16, rows x Kdim) * W(bf16, Nout x Kdim)^T + bias
// Block = 256 threads = 8 waves (4 M-waves x 2 N-waves). Wave tile 32x64.
// Block tile 128x128.  EPI: 0=f32 store, 1=sigmoid->bf16, 2=silu->bf16,
// 3=x1=resid+gate*acc (f32), 4=resid+acc (f32)
template <int EPI>
__global__ __launch_bounds__(256)
void gemm_xwt(const __bf16* __restrict__ A, const __bf16* __restrict__ W,
              const float* __restrict__ bias,
              const __bf16* __restrict__ gate, const float* __restrict__ resid,
              float* __restrict__ outF, __bf16* __restrict__ outB,
              int Kdim, int Nout) {
    const int lane = threadIdx.x & 31;
    const int wave = threadIdx.x >> 5;
    const int row0 = blockIdx.y * 128 + (wave >> 1) * 32;
    const int col0 = blockIdx.x * 128 + (wave & 1) * 64;

    v8f acc[2][4] = {};

    for (int k0 = 0; k0 < Kdim; k0 += 32) {
        if (k0 + 32 < Kdim)
            __builtin_prefetch(A + (size_t)(row0 + (lane & 15)) * Kdim + k0 + 32, 0, 1);
        v16bf a0 = load_fragA(A, Kdim, row0,      k0, lane);
        v16bf a1 = load_fragA(A, Kdim, row0 + 16, k0, lane);
        v16bf b0 = load_fragB(W, Kdim, col0,      k0, lane);
        v16bf b1 = load_fragB(W, Kdim, col0 + 16, k0, lane);
        v16bf b2 = load_fragB(W, Kdim, col0 + 32, k0, lane);
        v16bf b3 = load_fragB(W, Kdim, col0 + 48, k0, lane);
        acc[0][0] = __builtin_amdgcn_wmma_f32_16x16x32_bf16(false, a0, false, b0, (short)0, acc[0][0], false, false);
        acc[0][1] = __builtin_amdgcn_wmma_f32_16x16x32_bf16(false, a0, false, b1, (short)0, acc[0][1], false, false);
        acc[0][2] = __builtin_amdgcn_wmma_f32_16x16x32_bf16(false, a0, false, b2, (short)0, acc[0][2], false, false);
        acc[0][3] = __builtin_amdgcn_wmma_f32_16x16x32_bf16(false, a0, false, b3, (short)0, acc[0][3], false, false);
        acc[1][0] = __builtin_amdgcn_wmma_f32_16x16x32_bf16(false, a1, false, b0, (short)0, acc[1][0], false, false);
        acc[1][1] = __builtin_amdgcn_wmma_f32_16x16x32_bf16(false, a1, false, b1, (short)0, acc[1][1], false, false);
        acc[1][2] = __builtin_amdgcn_wmma_f32_16x16x32_bf16(false, a1, false, b2, (short)0, acc[1][2], false, false);
        acc[1][3] = __builtin_amdgcn_wmma_f32_16x16x32_bf16(false, a1, false, b3, (short)0, acc[1][3], false, false);
    }

    // C/D layout: lane%16 = col, VGPR r -> row r (lanes 0-15) / 8+r (lanes 16-31)
    const int cn = lane & 15;
    const int half = lane >> 4;
#pragma unroll
    for (int i = 0; i < 2; ++i) {
#pragma unroll
        for (int j = 0; j < 4; ++j) {
            const int cg = col0 + j * 16 + cn;
            const float bv = bias ? bias[cg] : 0.0f;
            const int rb = row0 + i * 16 + half * 8;
#pragma unroll
            for (int r = 0; r < 8; ++r) {
                const size_t off = (size_t)(rb + r) * Nout + cg;
                float v = acc[i][j][r] + bv;
                if constexpr (EPI == 0) {
                    outF[off] = v;
                } else if constexpr (EPI == 1) {
                    outB[off] = (__bf16)(1.0f / (1.0f + __expf(-v)));
                } else if constexpr (EPI == 2) {
                    outB[off] = (__bf16)(v / (1.0f + __expf(-v)));
                } else if constexpr (EPI == 3) {
                    outF[off] = resid[off] + (float)gate[off] * v;
                } else {
                    outF[off] = resid[off] + v;
                }
            }
        }
    }
}

// ---------------------------------------------------------------------------
__global__ __launch_bounds__(256)
void ln_to_bf16(const float* __restrict__ x, const float* __restrict__ g,
                const float* __restrict__ bta, __bf16* __restrict__ out) {
    __shared__ float red[256];
    const int tid = threadIdx.x;
    const size_t row = blockIdx.x;
    const float* xr = x + row * DDIM;
    float4 xv = *reinterpret_cast<const float4*>(xr + tid * 4);
    red[tid] = (xv.x + xv.y) + (xv.z + xv.w);
    __syncthreads();
#pragma unroll
    for (int off = 128; off > 0; off >>= 1) {
        if (tid < off) red[tid] += red[tid + off];
        __syncthreads();
    }
    const float mean = red[0] * (1.0f / DDIM);
    __syncthreads();
    const float d0 = xv.x - mean, d1 = xv.y - mean, d2 = xv.z - mean, d3 = xv.w - mean;
    red[tid] = d0 * d0 + d1 * d1 + d2 * d2 + d3 * d3;
    __syncthreads();
#pragma unroll
    for (int off = 128; off > 0; off >>= 1) {
        if (tid < off) red[tid] += red[tid + off];
        __syncthreads();
    }
    const float inv = rsqrtf(red[0] * (1.0f / DDIM) + 1e-5f);
    float4 gv = *reinterpret_cast<const float4*>(g + tid * 4);
    float4 bv = *reinterpret_cast<const float4*>(bta + tid * 4);
    __bf16* o = out + row * DDIM + tid * 4;
    o[0] = (__bf16)(d0 * inv * gv.x + bv.x);
    o[1] = (__bf16)(d1 * inv * gv.y + bv.y);
    o[2] = (__bf16)(d2 * inv * gv.z + bv.z);
    o[3] = (__bf16)(d3 * inv * gv.w + bv.w);
}

__global__ void cvt_f32_bf16(const float* __restrict__ s, __bf16* __restrict__ d, int n) {
    int i = blockIdx.x * blockDim.x + threadIdx.x;
    if (i < n) d[i] = (__bf16)s[i];
}

// From beta/delta rows build lambda (decay*e^{i freq}) and beta_c; c initialized to beta.
__global__ void prep_scan(const float* __restrict__ beta, const float* __restrict__ delta,
                          const float* __restrict__ logd, const float* __restrict__ freq0,
                          float* __restrict__ lamr, float* __restrict__ lami,
                          float* __restrict__ cr, float* __restrict__ ci, int total) {
    int i = blockIdx.x * blockDim.x + threadIdx.x;
    if (i >= total) return;
    const int k = i & (KCH - 1);
    const size_t tok = (size_t)(i >> 6);
    const float br = beta[tok * TWOK + k];
    const float bi = beta[tok * TWOK + KCH + k];
    const float dd = delta[tok * TWOK + k];
    const float df = delta[tok * TWOK + KCH + k];
    const float mag = 1.0f / (1.0f + __expf(-(logd[k] + 0.1f * tanhf(dd))));
    const float fr = freq0[k] + 0.05f * tanhf(df);
    float sn, cs;
    __sincosf(fr, &sn, &cs);
    lamr[i] = mag * cs;
    lami[i] = mag * sn;
    cr[i] = br;
    ci[i] = bi;
}

// Complex first-order scan c_t = lam_t*c_{t-1} + beta_t, one block per (b,k)
// sequence of length SEQT. 3-phase: serial-16 / block Hillis-Steele / apply.
__global__ __launch_bounds__(256)
void scan_kernel(const float* __restrict__ lamr, const float* __restrict__ lami,
                 float* __restrict__ cr, float* __restrict__ ci) {
    __shared__ float4 buf[2][256];
    const int seq = blockIdx.x;           // 0..B*KCH-1
    const int b = seq >> 6, k = seq & 63;
    const int tid = threadIdx.x;
    const int CH = SEQT / 256;            // 16

    float lr[16], li[16], br_[16], bi_[16];
    const size_t base = ((size_t)b * SEQT) * KCH + k;
#pragma unroll
    for (int e = 0; e < CH; ++e) {
        size_t idx = base + (size_t)(tid * CH + e) * KCH;
        lr[e] = lamr[idx]; li[e] = lami[idx];
        br_[e] = cr[idx];  bi_[e] = ci[idx];
    }
    // local serial scan; summary = (P = prod lam, c = local scan tail)
    float Pr = 1.0f, Pi = 0.0f, cvr = 0.0f, cvi = 0.0f;
#pragma unroll
    for (int e = 0; e < CH; ++e) {
        float nr = lr[e] * cvr - li[e] * cvi + br_[e];
        float ni = lr[e] * cvi + li[e] * cvr + bi_[e];
        cvr = nr; cvi = ni;
        float pr = lr[e] * Pr - li[e] * Pi;
        float pi = lr[e] * Pi + li[e] * Pr;
        Pr = pr; Pi = pi;
    }
    int src = 0;
    buf[0][tid] = make_float4(Pr, Pi, cvr, cvi);
    __syncthreads();
    for (int off = 1; off < 256; off <<= 1) {
        float4 cur = buf[src][tid];
        if (tid >= off) {
            float4 pre = buf[src][tid - off];
            float ar = cur.x * pre.x - cur.y * pre.y;
            float ai = cur.x * pre.y + cur.y * pre.x;
            float brr = cur.x * pre.z - cur.y * pre.w + cur.z;
            float bii = cur.x * pre.w + cur.y * pre.z + cur.w;
            cur = make_float4(ar, ai, brr, bii);
        }
        buf[src ^ 1][tid] = cur;
        __syncthreads();
        src ^= 1;
    }
    float p0r = 0.0f, p0i = 0.0f;
    if (tid > 0) { float4 p = buf[src][tid - 1]; p0r = p.z; p0i = p.w; }
    cvr = p0r; cvi = p0i;
#pragma unroll
    for (int e = 0; e < CH; ++e) {
        float nr = lr[e] * cvr - li[e] * cvi + br_[e];
        float ni = lr[e] * cvi + li[e] * cvr + bi_[e];
        cvr = nr; cvi = ni;
        size_t idx = base + (size_t)(tid * CH + e) * KCH;
        cr[idx] = cvr; ci[idx] = cvi;
    }
}

// 4 heads x 16x16 coupling applied to c_r/c_i, emit bf16 [cr', ci'] (token x 128)
__global__ __launch_bounds__(256)
void coupling_kernel(const float* __restrict__ cr, const float* __restrict__ ci,
                     const float* __restrict__ cb, __bf16* __restrict__ ccat) {
    __shared__ float W[4 * 16 * 16];
    for (int i = threadIdx.x; i < 1024; i += 256) W[i] = cb[i];
    __syncthreads();
    const size_t t = (size_t)blockIdx.x * 4 + (threadIdx.x >> 6);
    const int j = threadIdx.x & 63;
    const int h = j >> 4, jj = j & 15;
    const float* rb = cr + t * KCH + h * 16;
    const float* ib = ci + t * KCH + h * 16;
    const float* wr = &W[(h * 16 + jj) * 16];
    float sr = 0.0f, si = 0.0f;
#pragma unroll
    for (int kk = 0; kk < 16; ++kk) { sr += wr[kk] * rb[kk]; si += wr[kk] * ib[kk]; }
    ccat[t * TWOK + j] = (__bf16)sr;
    ccat[t * TWOK + KCH + j] = (__bf16)si;
}

// ---------------------------------------------------------------------------
extern "C" void kernel_launch(void* const* d_in, const int* in_sizes, int n_in,
                              void* d_out, int out_size, void* d_ws, size_t ws_size,
                              hipStream_t stream) {
    (void)in_sizes; (void)n_in; (void)out_size; (void)ws_size;
    const float* x          = (const float*)d_in[0];
    const float* in_proj_w  = (const float*)d_in[1];
    const float* log_decay  = (const float*)d_in[2];
    const float* frequency  = (const float*)d_in[3];
    const float* select_w   = (const float*)d_in[4];
    const float* select_b   = (const float*)d_in[5];
    const float* coupling   = (const float*)d_in[6];
    const float* out_proj_w = (const float*)d_in[7];
    const float* gate_w     = (const float*)d_in[8];
    const float* gate_b     = (const float*)d_in[9];
    const float* mlp_w1     = (const float*)d_in[10];
    const float* mlp_b1     = (const float*)d_in[11];
    const float* mlp_w2     = (const float*)d_in[12];
    const float* mlp_b2     = (const float*)d_in[13];
    const float* ln1_g      = (const float*)d_in[14];
    const float* ln1_b      = (const float*)d_in[15];
    const float* ln2_g      = (const float*)d_in[16];
    const float* ln2_b      = (const float*)d_in[17];
    float* out = (float*)d_out;

    char* p = (char*)d_ws;
    auto alloc = [&](size_t bytes) -> char* {
        char* r = p;
        p += (bytes + 255) & ~(size_t)255;
        return r;
    };
    __bf16* xn_bf   = (__bf16*)alloc((size_t)NTOK * DDIM * 2);
    __bf16* x2_bf   = (__bf16*)alloc((size_t)NTOK * DDIM * 2);
    __bf16* w_in    = (__bf16*)alloc((size_t)TWOK * DDIM * 2);
    __bf16* w_sel   = (__bf16*)alloc((size_t)TWOK * DDIM * 2);
    __bf16* w_gate  = (__bf16*)alloc((size_t)DDIM * DDIM * 2);
    __bf16* w_out   = (__bf16*)alloc((size_t)DDIM * TWOK * 2);
    __bf16* w_m1    = (__bf16*)alloc((size_t)MLPD * DDIM * 2);
    __bf16* w_m2    = (__bf16*)alloc((size_t)DDIM * MLPD * 2);
    float*  beta    = (float*)alloc((size_t)NTOK * TWOK * 4);
    float*  delta   = (float*)alloc((size_t)NTOK * TWOK * 4);
    __bf16* gate_bf = (__bf16*)alloc((size_t)NTOK * DDIM * 2);
    float*  lamr    = (float*)alloc((size_t)NTOK * KCH * 4);
    float*  lami    = (float*)alloc((size_t)NTOK * KCH * 4);
    float*  cre     = (float*)alloc((size_t)NTOK * KCH * 4);
    float*  cim     = (float*)alloc((size_t)NTOK * KCH * 4);
    __bf16* ccat    = (__bf16*)alloc((size_t)NTOK * TWOK * 2);
    float*  x1      = (float*)alloc((size_t)NTOK * DDIM * 4);
    __bf16* m_bf    = (__bf16*)alloc((size_t)NTOK * MLPD * 2);

    // LN1
    ln_to_bf16<<<NTOK, 256, 0, stream>>>(x, ln1_g, ln1_b, xn_bf);
    // weight conversions
    cvt_f32_bf16<<<(TWOK * DDIM + 255) / 256, 256, 0, stream>>>(in_proj_w, w_in, TWOK * DDIM);
    cvt_f32_bf16<<<(TWOK * DDIM + 255) / 256, 256, 0, stream>>>(select_w, w_sel, TWOK * DDIM);
    cvt_f32_bf16<<<(DDIM * DDIM + 255) / 256, 256, 0, stream>>>(gate_w, w_gate, DDIM * DDIM);
    cvt_f32_bf16<<<(DDIM * TWOK + 255) / 256, 256, 0, stream>>>(out_proj_w, w_out, DDIM * TWOK);
    cvt_f32_bf16<<<(MLPD * DDIM + 255) / 256, 256, 0, stream>>>(mlp_w1, w_m1, MLPD * DDIM);
    cvt_f32_bf16<<<(DDIM * MLPD + 255) / 256, 256, 0, stream>>>(mlp_w2, w_m2, DDIM * MLPD);

    const dim3 blk(256);
    // beta = xn @ in_proj_w^T
    gemm_xwt<0><<<dim3(TWOK / 128, NTOK / 128), blk, 0, stream>>>(
        xn_bf, w_in, nullptr, nullptr, nullptr, beta, nullptr, DDIM, TWOK);
    // delta = xn @ select_w^T + select_b
    gemm_xwt<0><<<dim3(TWOK / 128, NTOK / 128), blk, 0, stream>>>(
        xn_bf, w_sel, select_b, nullptr, nullptr, delta, nullptr, DDIM, TWOK);
    // gate = sigmoid(xn @ gate_w^T + gate_b) -> bf16
    gemm_xwt<1><<<dim3(DDIM / 128, NTOK / 128), blk, 0, stream>>>(
        xn_bf, w_gate, gate_b, nullptr, nullptr, nullptr, gate_bf, DDIM, DDIM);

    // lambda / beta_c
    prep_scan<<<(NTOK * KCH + 255) / 256, 256, 0, stream>>>(
        beta, delta, log_decay, frequency, lamr, lami, cre, cim, NTOK * KCH);
    // associative complex scan along T per (b,k)
    scan_kernel<<<NB * KCH, 256, 0, stream>>>(lamr, lami, cre, cim);
    // head coupling, pack c_cat bf16
    coupling_kernel<<<NTOK / 4, 256, 0, stream>>>(cre, cim, coupling, ccat);

    // x1 = x + gate * (c_cat @ out_proj_w^T)
    gemm_xwt<3><<<dim3(DDIM / 128, NTOK / 128), blk, 0, stream>>>(
        ccat, w_out, nullptr, gate_bf, x, x1, nullptr, TWOK, DDIM);
    // LN2
    ln_to_bf16<<<NTOK, 256, 0, stream>>>(x1, ln2_g, ln2_b, x2_bf);
    // m = silu(x2 @ mlp_w1^T + b1) -> bf16
    gemm_xwt<2><<<dim3(MLPD / 128, NTOK / 128), blk, 0, stream>>>(
        x2_bf, w_m1, mlp_b1, nullptr, nullptr, nullptr, m_bf, DDIM, MLPD);
    // out = x1 + m @ mlp_w2^T + b2
    gemm_xwt<4><<<dim3(DDIM / 128, NTOK / 128), blk, 0, stream>>>(
        m_bf, w_m2, mlp_b2, nullptr, x1, out, nullptr, MLPD, DDIM);
}
